// WSWGAT_10093173145792
// MI455X (gfx1250) — matrix operand order repaired and checked by reading, not compiled, not run on hardware
//
#include <hip/hip_runtime.h>
#include <math.h>

typedef __attribute__((ext_vector_type(16))) _Float16 v16h;
typedef __attribute__((ext_vector_type(8)))  _Float16 v8h;
typedef __attribute__((ext_vector_type(4)))  _Float16 v4h;
typedef __attribute__((ext_vector_type(8)))  float    v8f;

#define H_HEADS 8
#define DHD     16
#define DIM     128
#define FFND    512

// ---------------- WMMA helpers (wave32, v_wmma_f32_16x16x32_f16) ----------------

__device__ __forceinline__ v8f wmma_f16(v16h a, v16h b, v8f c) {
  return __builtin_amdgcn_wmma_f32_16x16x32_f16(false, a, false, b, (short)0, c,
                                                false, false);
}

// A fragment 16x32 f16 from row-major f16: lane's data = two contiguous 16B runs
// (K = half*8..+8 and K = 16+half*8..+8).  p0 must be 16B aligned.
__device__ __forceinline__ v16h load_A8(const _Float16* p0) {
  v8h lo = *(const v8h*)p0;
  v8h hi = *(const v8h*)(p0 + 16);
  v16h a;
#pragma unroll
  for (int v = 0; v < 8; ++v) { a[v] = lo[v]; a[v + 8] = hi[v]; }
  return a;
}

// ---------------- Kernel 0: init accumulators ----------------
__global__ void init_kernel(float* neigh, float* m, float* denom, int n_nodes) {
  int i = blockIdx.x * blockDim.x + threadIdx.x;
  if (i < n_nodes * DIM) neigh[i] = 0.0f;
  if (i < n_nodes * H_HEADS) { m[i] = -INFINITY; denom[i] = 0.0f; }
}

// ---------------- weight pre-passes (run once, tiny) ----------------

// f32 -> f16 row-major (node features), 4 elements/thread
__global__ void convert_w_kernel(const float* __restrict__ w, _Float16* __restrict__ wh,
                                 int total4) {
  int i = blockIdx.x * blockDim.x + threadIdx.x;
  if (i >= total4) return;
  float4 f = ((const float4*)w)[i];
  v4h o; o[0] = (_Float16)f.x; o[1] = (_Float16)f.y; o[2] = (_Float16)f.z; o[3] = (_Float16)f.w;
  ((v4h*)wh)[i] = o;
}

// Swizzle a KxN f32 row-major matrix into per-lane-contiguous f16 B fragments:
// dst[(tile*32 + lane)*16 + e].  nmajor=1: tile = n_tile*numK + k_tile; else
// tile = k_tile*numN + n_tile.
__global__ void swizzleB_kernel(const float* __restrict__ src, _Float16* __restrict__ dst,
                                int Ktot, int Ntot, int nmajor) {
  int i = blockIdx.x * blockDim.x + threadIdx.x;
  int numK = Ktot >> 5, numN = Ntot >> 4;
  if (i >= numK * numN * 32) return;
  int lane = i & 31, tile = i >> 5;
  int k_tile, n_tile;
  if (nmajor) { k_tile = tile % numK; n_tile = tile / numK; }
  else        { n_tile = tile % numN; k_tile = tile / numN; }
  int n = lane & 15, half = lane >> 4;
  v16h b;
#pragma unroll
  for (int v = 0; v < 8; ++v) {
    int k = half * 8 + 2 * v + (v >= 4 ? 8 : 0);
    b[2 * v]     = (_Float16)src[(size_t)(k_tile * 32 + k) * Ntot + n_tile * 16 + n];
    b[2 * v + 1] = (_Float16)src[(size_t)(k_tile * 32 + k + 1) * Ntot + n_tile * 16 + n];
  }
  *(v16h*)(dst + (size_t)(tile * 32 + lane) * 16) = b;
}

// ---------------- Kernel 1: z = w @ Wfc (wave owns 16 rows, loops 8 heads) ----
__global__ __launch_bounds__(256) void proj_kernel(const _Float16* __restrict__ wh,
                                                   const _Float16* __restrict__ Wfch,
                                                   float* __restrict__ z, int n_nodes) {
  int wave = threadIdx.x >> 5, lane = threadIdx.x & 31;
  int n0 = (blockIdx.x * 8 + wave) * 16;
  if (n0 >= n_nodes) return;                 // wave-uniform, no barriers here
  int M = lane & 15, half = lane >> 4;
  int row = n0 + M; if (row >= n_nodes) row = n_nodes - 1;
  v16h A[4];
#pragma unroll
  for (int kk = 0; kk < 4; ++kk)
    A[kk] = load_A8(wh + (size_t)row * DIM + kk * 32 + half * 8);

  int nloc = lane & 15, rbase = half * 8;
  bool full = (n0 + 16 <= n_nodes);          // wave-uniform fast path
#pragma unroll
  for (int h = 0; h < H_HEADS; ++h) {
    v8f acc = {};
#pragma unroll
    for (int kk = 0; kk < 4; ++kk) {
      v16h b = *(const v16h*)(Wfch + (size_t)((h * 4 + kk) * 32 + lane) * 16);
      acc = wmma_f16(A[kk], b, acc);
    }
    int col = h * DHD + nloc;
    float* zp = z + (size_t)(n0 + rbase) * DIM + col;
    if (full) {
#pragma unroll
      for (int v = 0; v < 8; ++v) zp[(size_t)v * DIM] = acc[v];
    } else {
#pragma unroll
      for (int v = 0; v < 8; ++v)
        if (n0 + rbase + v < n_nodes) zp[(size_t)v * DIM] = acc[v];
    }
  }
}

// ---------------- Kernel 2: per-node attention logit halves ----------------
__global__ void score_kernel(const float* __restrict__ z,
                             const float* __restrict__ a_src,
                             const float* __restrict__ a_dst,
                             float* es, float* ed, int n_nodes) {
  int i = blockIdx.x * blockDim.x + threadIdx.x;
  if (i >= n_nodes * H_HEADS) return;
  int n = i >> 3, h = i & 7;
  const float* zp = z + (size_t)n * DIM + h * DHD;
  const float* as = a_src + h * DHD;
  const float* ad = a_dst + h * DHD;
  float s1 = 0.0f, s2 = 0.0f;
#pragma unroll
  for (int d = 0; d < DHD; ++d) { float v = zp[d]; s1 += v * as[d]; s2 += v * ad[d]; }
  es[i] = s1; ed[i] = s2;
}

__device__ __forceinline__ float leaky(float x) { return x > 0.0f ? x : 0.01f * x; }

// ---------------- Kernel 3: segment max (bit-trick float atomic max) --------
__global__ void edge_max_kernel(const int* __restrict__ src, const int* __restrict__ dst,
                                const float* __restrict__ es, const float* __restrict__ ed,
                                float* m, int n_edges) {
  int i = blockIdx.x * blockDim.x + threadIdx.x;
  if (i >= n_edges * H_HEADS) return;
  int e = i >> 3, h = i & 7;
  int sN = src[e], dN = dst[e];
  float sc = leaky(es[sN * H_HEADS + h] + ed[dN * H_HEADS + h]);
  float* addr = m + dN * H_HEADS + h;
  if (sc >= 0.0f) atomicMax((int*)addr, __float_as_int(sc));
  else            atomicMin((unsigned int*)addr, __float_as_uint(sc));
}

// ---------------- Kernel 4: segment sum of exp ----------------
__global__ void edge_sum_kernel(const int* __restrict__ src, const int* __restrict__ dst,
                                const float* __restrict__ es, const float* __restrict__ ed,
                                const float* __restrict__ m, float* denom, int n_edges) {
  int i = blockIdx.x * blockDim.x + threadIdx.x;
  if (i >= n_edges * H_HEADS) return;
  int e = i >> 3, h = i & 7;
  int sN = src[e], dN = dst[e];
  float sc = leaky(es[sN * H_HEADS + h] + ed[dN * H_HEADS + h]);
  atomicAdd(denom + dN * H_HEADS + h, expf(sc - m[dN * H_HEADS + h]));
}

// ---------------- Kernel 5: alpha-weighted message scatter (L2-resident) ----
__global__ void edge_msg_kernel(const int* __restrict__ src, const int* __restrict__ dst,
                                const float* __restrict__ es, const float* __restrict__ ed,
                                const float* __restrict__ m, const float* __restrict__ denom,
                                const float* __restrict__ z, float* neigh, int n_edges) {
  int i = blockIdx.x * blockDim.x + threadIdx.x;
  if (i >= n_edges * H_HEADS) return;
  int e = i >> 3, h = i & 7;
  int sN = src[e], dN = dst[e];
  float sc = leaky(es[sN * H_HEADS + h] + ed[dN * H_HEADS + h]);
  float alpha = expf(sc - m[dN * H_HEADS + h]) / (denom[dN * H_HEADS + h] + 1e-9f);
  const float4* zr = (const float4*)(z + (size_t)sN * DIM + h * DHD);
  float* np = neigh + (size_t)dN * DIM + h * DHD;
#pragma unroll
  for (int q = 0; q < 4; ++q) {
    float4 zz = zr[q];
    atomicAdd(np + 4 * q + 0, alpha * zz.x);
    atomicAdd(np + 4 * q + 1, alpha * zz.y);
    atomicAdd(np + 4 * q + 2, alpha * zz.z);
    atomicAdd(np + 4 * q + 3, alpha * zz.w);
  }
}

// ---------------- Kernel 6: fused ELU+LN+FFN+residual ----------------------
// Wave owns 16 nodes.  Per 32-wide FFN super-chunk, the block stages 16KB of
// pre-swizzled w1/w2 fragments into LDS once via global_load_async_to_lds_b128
// (shared by all 8 waves), then GEMM1 -> ReLU -> LDS relayout -> GEMM2.
__global__ __launch_bounds__(256) void ffn_kernel(const float* __restrict__ s,
                                                  const float* __restrict__ neigh,
                                                  const float* __restrict__ ln_g,
                                                  const float* __restrict__ ln_b,
                                                  const _Float16* __restrict__ w1h,
                                                  const float* __restrict__ b1,
                                                  const _Float16* __restrict__ w2h,
                                                  const float* __restrict__ b2,
                                                  float* __restrict__ out, int n_nodes) {
  __shared__ __align__(32) _Float16 xnS[8][16][DIM + 8];  // 34.8 KB
  __shared__ __align__(32) _Float16 TS[8][16][40];        // 10.2 KB
  __shared__ __align__(32) _Float16 Wstage[8192];         // 16 KB: [0,4096)=w1, [4096,8192)=w2

  int wave = threadIdx.x >> 5;
  int lane = threadIdx.x & 31;
  int n0 = (blockIdx.x * 8 + wave) * 16;
  if (n0 >= n_nodes) n0 = ((n_nodes - 1) >> 4) << 4;  // clamp: keep waves alive for barriers

  // ---- ELU + LayerNorm: 2 lanes per row, 64 cols each ----
  int r = lane >> 1, part = lane & 1;
  int n = n0 + r; if (n > n_nodes - 1) n = n_nodes - 1;
  const float* sp = s + (size_t)n * DIM;
  const float* gp = neigh + (size_t)n * DIM;
  float xv[64];
  float sum = 0.0f, ss = 0.0f;
#pragma unroll
  for (int c = 0; c < 64; ++c) {
    int col = part * 64 + c;
    float x = sp[col] + gp[col];
    x = x > 0.0f ? x : (expf(x) - 1.0f);   // ELU(alpha=1)
    xv[c] = x; sum += x; ss += x * x;
  }
  sum += __shfl_xor(sum, 1, 32);
  ss  += __shfl_xor(ss, 1, 32);
  float mu   = sum * (1.0f / DIM);
  float var  = ss * (1.0f / DIM) - mu * mu;
  float rstd = rsqrtf(var + 1e-6f);
#pragma unroll
  for (int c = 0; c < 64; ++c) {
    int col = part * 64 + c;
    xnS[wave][r][col] = (_Float16)((xv[c] - mu) * rstd * ln_g[col] + ln_b[col]);
  }
  asm volatile("s_wait_dscnt 0x0" ::: "memory");

  int M = lane & 15, half = lane >> 4;
  v16h Ax[4];
#pragma unroll
  for (int kk = 0; kk < 4; ++kk)
    Ax[kk] = load_A8(&xnS[wave][M][kk * 32 + half * 8]);

  v8f acc[8] = {};
  int nloc = lane & 15;
  int rbase = half * 8;
  const char* w1b = (const char*)w1h;
  const char* w2b = (const char*)w2h;
  unsigned lds0 = (unsigned)(uintptr_t)&Wstage[0];

  for (int scc = 0; scc < 16; ++scc) {     // 16 super-chunks of 32 FFN dims
    __syncthreads();                       // everyone done reading previous stage
    {                                      // cooperative async DMA: 16KB -> LDS
      unsigned toff = (unsigned)threadIdx.x * 16u;
      unsigned long long g1 = (unsigned long long)(uintptr_t)(w1b + scc * 8192 + toff);
      unsigned long long g2 = (unsigned long long)(uintptr_t)(w2b + scc * 8192 + toff);
      asm volatile("global_load_async_to_lds_b128 %0, %1, off"
                   :: "v"(lds0 + toff),          "v"(g1)          : "memory");
      asm volatile("global_load_async_to_lds_b128 %0, %1, off"
                   :: "v"(lds0 + 4096u + toff),  "v"(g1 + 4096ull): "memory");
      asm volatile("global_load_async_to_lds_b128 %0, %1, off"
                   :: "v"(lds0 + 8192u + toff),  "v"(g2)          : "memory");
      asm volatile("global_load_async_to_lds_b128 %0, %1, off"
                   :: "v"(lds0 + 12288u + toff), "v"(g2 + 4096ull): "memory");
      asm volatile("s_wait_asynccnt 0x0" ::: "memory");
    }
    __syncthreads();                       // stage visible to all waves

#pragma unroll
    for (int t = 0; t < 2; ++t) {          // two 16-wide FFN col tiles
      int j = scc * 2 + t;
      v8f d = {};
#pragma unroll
      for (int kk = 0; kk < 4; ++kk) {
        v16h b = *(const v16h*)(&Wstage[(size_t)((t * 4 + kk) * 32 + lane) * 16]);
        d = wmma_f16(Ax[kk], b, d);
      }
      float b1v = b1[j * 16 + nloc];
#pragma unroll
      for (int v = 0; v < 8; ++v) {
        float val = d[v] + b1v;
        val = val > 0.0f ? val : 0.0f;     // ReLU
        TS[wave][rbase + v][t * 16 + nloc] = (_Float16)val;
      }
    }
    asm volatile("s_wait_dscnt 0x0" ::: "memory");
    v16h At = load_A8(&TS[wave][M][half * 8]);
#pragma unroll
    for (int ct = 0; ct < 8; ++ct) {
      v16h b = *(const v16h*)(&Wstage[4096 + (size_t)(ct * 32 + lane) * 16]);
      acc[ct] = wmma_f16(At, b, acc[ct]);
    }
  }

  // ---- epilogue: + b2 + residual ELU(s + neigh), uniform fast path ----
  bool full = (n0 + 16 <= n_nodes);
#pragma unroll
  for (int ct = 0; ct < 8; ++ct) {
    int col = ct * 16 + nloc;
    float b2v = b2[col];
    const float* sp2 = s + (size_t)(n0 + rbase) * DIM + col;
    const float* gp2 = neigh + (size_t)(n0 + rbase) * DIM + col;
    float* op = out + (size_t)(n0 + rbase) * DIM + col;
    if (full) {
#pragma unroll
      for (int v = 0; v < 8; ++v) {
        float x = sp2[(size_t)v * DIM] + gp2[(size_t)v * DIM];
        x = x > 0.0f ? x : (expf(x) - 1.0f);
        op[(size_t)v * DIM] = acc[ct][v] + b2v + x;
      }
    } else {
#pragma unroll
      for (int v = 0; v < 8; ++v) {
        if (n0 + rbase + v < n_nodes) {
          float x = sp2[(size_t)v * DIM] + gp2[(size_t)v * DIM];
          x = x > 0.0f ? x : (expf(x) - 1.0f);
          op[(size_t)v * DIM] = acc[ct][v] + b2v + x;
        }
      }
    }
  }
}

// ---------------- launch ----------------
extern "C" void kernel_launch(void* const* d_in, const int* in_sizes, int n_in,
                              void* d_out, int out_size, void* d_ws, size_t ws_size,
                              hipStream_t stream) {
  const float* w     = (const float*)d_in[0];
  const float* s     = (const float*)d_in[1];
  const int*   src   = (const int*)d_in[2];
  const int*   dst   = (const int*)d_in[3];
  const float* Wfc   = (const float*)d_in[4];
  const float* a_src = (const float*)d_in[5];
  const float* a_dst = (const float*)d_in[6];
  const float* ln_g  = (const float*)d_in[7];
  const float* ln_b  = (const float*)d_in[8];
  const float* w1    = (const float*)d_in[9];
  const float* b1    = (const float*)d_in[10];
  const float* w2    = (const float*)d_in[11];
  const float* b2    = (const float*)d_in[12];
  float* out = (float*)d_out;

  int n_nodes = in_sizes[0] / DIM;
  int n_edges = in_sizes[2];

  char* wsb = (char*)d_ws;
  float* z      = (float*)wsb; wsb += (size_t)n_nodes * DIM * sizeof(float);
  float* neigh  = (float*)wsb; wsb += (size_t)n_nodes * DIM * sizeof(float);
  float* es     = (float*)wsb; wsb += (size_t)n_nodes * H_HEADS * sizeof(float);
  float* ed     = (float*)wsb; wsb += (size_t)n_nodes * H_HEADS * sizeof(float);
  float* m      = (float*)wsb; wsb += (size_t)n_nodes * H_HEADS * sizeof(float);
  float* denom  = (float*)wsb; wsb += (size_t)n_nodes * H_HEADS * sizeof(float);
  _Float16* wh   = (_Float16*)wsb; wsb += (size_t)n_nodes * DIM * sizeof(_Float16);
  _Float16* Wfch = (_Float16*)wsb; wsb += (size_t)H_HEADS * 4 * 32 * 16 * sizeof(_Float16);
  _Float16* w1h  = (_Float16*)wsb; wsb += (size_t)(DIM / 32) * (FFND / 16) * 512 * sizeof(_Float16);
  _Float16* w2h  = (_Float16*)wsb;

  dim3 blk(256);
  init_kernel<<<(n_nodes * DIM + 255) / 256, blk, 0, stream>>>(neigh, m, denom, n_nodes);
  convert_w_kernel<<<(n_nodes * DIM / 4 + 255) / 256, blk, 0, stream>>>(w, wh, n_nodes * DIM / 4);
  swizzleB_kernel<<<16, blk, 0, stream>>>(w1, w1h, DIM, FFND, 1);   // tiles j*4+kk
  swizzleB_kernel<<<16, blk, 0, stream>>>(w2, w2h, FFND, DIM, 0);   // tiles scc*8+ct
  for (int h = 0; h < H_HEADS; ++h)
    swizzleB_kernel<<<1, blk, 0, stream>>>(Wfc + (size_t)h * DIM * DHD,
                                           Wfch + (size_t)h * 4 * 512, DIM, DHD, 1);

  proj_kernel<<<(n_nodes + 127) / 128, blk, 0, stream>>>(wh, Wfch, z, n_nodes);
  score_kernel<<<(n_nodes * H_HEADS + 255) / 256, blk, 0, stream>>>(z, a_src, a_dst,
                                                                    es, ed, n_nodes);
  int egrid = (n_edges * H_HEADS + 255) / 256;
  edge_max_kernel<<<egrid, blk, 0, stream>>>(src, dst, es, ed, m, n_edges);
  edge_sum_kernel<<<egrid, blk, 0, stream>>>(src, dst, es, ed, m, denom, n_edges);
  edge_msg_kernel<<<egrid, blk, 0, stream>>>(src, dst, es, ed, m, denom, z, neigh, n_edges);
  ffn_kernel<<<(n_nodes + 127) / 128, blk, 0, stream>>>(s, neigh, ln_g, ln_b,
                                                        w1h, b1, w2h, b2, out, n_nodes);
}
